// DeepFRIModel_17188459119373
// MI455X (gfx1250) — compile-verified, matrix operand order checked by reading, not run on hardware
//
#include <hip/hip_runtime.h>
#include <hip/hip_bf16.h>
#include <hip/hip_fp16.h>

// ---------------- problem constants (match reference) ----------------
#define N_NODES 50000
#define N_EDGES 800000
#define IN_DIMC 1280
#define GDIM    512
#define CATDIM  1536
#define FC0D    1024
#define FC1D    512
#define OUTD    256
#define NGRAPH  64

typedef __attribute__((ext_vector_type(16))) _Float16 v16h;
typedef __attribute__((ext_vector_type(8)))  float    v8f;

// ---------------------------------------------------------------------
// Generic WMMA GEMM:  C[M][Nout] = A[M][K] @ Bt[Nout][K]^T  (+bias, relu)
// A, Bt are f16; accumulation fp32 via v_wmma_f32_16x16x32_f16.
// One wave computes one 16x16 output tile. M % 16 == 0, K % 32 == 0,
// Nout % 16 == 0 (all true for this model).
// Fragment layout per CDNA5 ISA 7.12.2 (16-bit A 16x32; B symmetric):
//   lane = 32 threads; m/n = lane&15; half = lane>>4
//   elem j (j<8):  k = kbase + half*8 + j
//   elem j (j>=8): k = kbase + 16 + half*8 + (j-8)
// ---------------------------------------------------------------------
__global__ __launch_bounds__(256)
void gemm_f16_wmma(const _Float16* __restrict__ A,
                   const _Float16* __restrict__ Bt,
                   const float* __restrict__ bias,   // may be null
                   float* __restrict__ Cf32,         // may be null
                   _Float16* __restrict__ Cf16,      // may be null
                   int M, int K, int Nout, int relu)
{
    const int lane  = threadIdx.x & 31;
    const int wave  = threadIdx.x >> 5;
    const int ntile = Nout >> 4;
    const int tiles = (M >> 4) * ntile;
    const int tile  = blockIdx.x * 8 + wave;
    if (tile >= tiles) return;

    const int mt   = tile / ntile;
    const int nt   = tile % ntile;
    const int half = lane >> 4;
    const int l15  = lane & 15;

    const _Float16* __restrict__ arow = A  + (size_t)((mt << 4) + l15) * K;
    const _Float16* __restrict__ brow = Bt + (size_t)((nt << 4) + l15) * K;

    v8f c = {};
    for (int k = 0; k < K; k += 32) {
        const int k0 = k + half * 8;
        const int k1 = k + 16 + half * 8;
        uint4 ua0 = *reinterpret_cast<const uint4*>(arow + k0);
        uint4 ua1 = *reinterpret_cast<const uint4*>(arow + k1);
        uint4 ub0 = *reinterpret_cast<const uint4*>(brow + k0);
        uint4 ub1 = *reinterpret_cast<const uint4*>(brow + k1);
        const _Float16* pa0 = reinterpret_cast<const _Float16*>(&ua0);
        const _Float16* pa1 = reinterpret_cast<const _Float16*>(&ua1);
        const _Float16* pb0 = reinterpret_cast<const _Float16*>(&ub0);
        const _Float16* pb1 = reinterpret_cast<const _Float16*>(&ub1);
        v16h a, b;
#pragma unroll
        for (int j = 0; j < 8; ++j) {
            a[j]     = pa0[j];
            a[j + 8] = pa1[j];
            b[j]     = pb0[j];
            b[j + 8] = pb1[j];
        }
        c = __builtin_amdgcn_wmma_f32_16x16x32_f16(
                false, a, false, b, (short)0, c, false, false);
    }

    const int ncol = (nt << 4) + l15;
    const float bval = bias ? bias[ncol] : 0.0f;
#pragma unroll
    for (int r = 0; r < 8; ++r) {
        const int m = (mt << 4) + half * 8 + r;   // C/D layout: VGPR r -> row half*8+r
        float v = c[r] + bval;
        if (relu) v = fmaxf(v, 0.0f);
        const size_t o = (size_t)m * Nout + ncol;
        if (Cf32) Cf32[o] = v;
        if (Cf16) Cf16[o] = (_Float16)v;
    }
}

// ------------------------- helper kernels ---------------------------
__global__ void cast_f32_to_f16(const float* __restrict__ x,
                                _Float16* __restrict__ y, int n)
{
    int i = blockIdx.x * blockDim.x + threadIdx.x;
    if (i < n) y[i] = (_Float16)x[i];
}

// W[K][Nout] (row-major) -> Wt[Nout][K] f16
__global__ void transpose_cast(const float* __restrict__ W,
                               _Float16* __restrict__ Wt, int K, int Nout)
{
    int i = blockIdx.x * blockDim.x + threadIdx.x;
    if (i >= K * Nout) return;
    int k = i / Nout, n = i % Nout;
    Wt[(size_t)n * K + k] = (_Float16)W[i];
}

__global__ void deg_init(float* __restrict__ deg, int n)
{
    int i = blockIdx.x * blockDim.x + threadIdx.x;
    if (i < n) deg[i] = 1.0f;                     // self loop
}

__global__ void deg_edges(const int* __restrict__ dst, float* __restrict__ deg, int e)
{
    int i = blockIdx.x * blockDim.x + threadIdx.x;
    if (i < e) unsafeAtomicAdd(&deg[dst[i]], 1.0f);
}

__global__ void deg_rsqrt(float* __restrict__ deg, int n)   // in-place -> dinv
{
    int i = blockIdx.x * blockDim.x + threadIdx.x;
    if (i < n) deg[i] = rsqrtf(deg[i]);
}

// agg[dst] += h[src] * dinv[src]*dinv[dst] ; one thread = (edge, 4 dims)
__global__ __launch_bounds__(256)
void edge_scatter(const int* __restrict__ src, const int* __restrict__ dst,
                  const float* __restrict__ dinv, const float* __restrict__ h,
                  float* __restrict__ agg)
{
    unsigned idx = blockIdx.x * blockDim.x + threadIdx.x;
    unsigned e = idx >> 7;                 // 128 quads of 4 dims = 512
    if (e >= N_EDGES) return;
    int q = (idx & 127) << 2;
    int s = src[e], d = dst[e];
    float norm = dinv[s] * dinv[d];
    float4 v = *reinterpret_cast<const float4*>(h + (size_t)s * GDIM + q);
    float* base = agg + (size_t)d * GDIM + q;
    unsafeAtomicAdd(base + 0, v.x * norm);
    unsafeAtomicAdd(base + 1, v.y * norm);
    unsafeAtomicAdd(base + 2, v.z * norm);
    unsafeAtomicAdd(base + 3, v.w * norm);
}

// h_out = agg + h*dinv^2 + bias ; in-place over h, optional f16 copy
__global__ void gcn_finalize(const float* __restrict__ agg,
                             float* __restrict__ h,            // in: XW, out: layer out
                             const float* __restrict__ bias,
                             const float* __restrict__ dinv,
                             _Float16* __restrict__ hf16)      // may be null
{
    unsigned i = blockIdx.x * blockDim.x + threadIdx.x;
    if (i >= (unsigned)N_NODES * GDIM) return;
    int node = i >> 9, d = i & 511;
    float di = dinv[node];
    float v = agg[i] + h[i] * di * di + bias[d];
    h[i] = v;
    if (hf16) hf16[i] = (_Float16)v;
}

// pooled sum: batch is sorted -> accumulate runs of 8 nodes before atomic
__global__ void pool_sum(const float* __restrict__ h, const int* __restrict__ batch,
                         float* __restrict__ g32, int off)
{
    unsigned idx = blockIdx.x * blockDim.x + threadIdx.x;
    const int chunks = N_NODES / 8;          // 6250 exactly
    if (idx >= (unsigned)chunks * GDIM) return;
    int d = idx & 511;
    int i0 = (idx >> 9) * 8;
    int cur = batch[i0];
    float acc = 0.0f;
#pragma unroll
    for (int i = i0; i < i0 + 8; ++i) {
        int g = batch[i];
        if (g != cur) {
            unsafeAtomicAdd(&g32[(size_t)cur * CATDIM + off + d], acc);
            acc = 0.0f; cur = g;
        }
        acc += h[(size_t)i * GDIM + d];
    }
    unsafeAtomicAdd(&g32[(size_t)cur * CATDIM + off + d], acc);
}

__global__ void count_nodes(const int* __restrict__ batch, float* __restrict__ cnt, int n)
{
    int i = blockIdx.x * blockDim.x + threadIdx.x;
    if (i < n) unsafeAtomicAdd(&cnt[batch[i]], 1.0f);
}

__global__ void pool_div(const float* __restrict__ cnt,
                         float* __restrict__ g32, _Float16* __restrict__ g16)
{
    int i = blockIdx.x * blockDim.x + threadIdx.x;
    if (i >= NGRAPH * CATDIM) return;
    float c = fmaxf(cnt[i / CATDIM], 1.0f);
    float v = g32[i] / c;
    g32[i] = v;
    g16[i] = (_Float16)v;
}

// --------------------------- launcher -------------------------------
extern "C" void kernel_launch(void* const* d_in, const int* in_sizes, int n_in,
                              void* d_out, int out_size, void* d_ws, size_t ws_size,
                              hipStream_t stream)
{
    const float* x     = (const float*)d_in[0];
    const int*   ei    = (const int*)d_in[1];
    const int*   batch = (const int*)d_in[2];
    const float* Wg1 = (const float*)d_in[3];  const float* bg1 = (const float*)d_in[4];
    const float* Wg2 = (const float*)d_in[5];  const float* bg2 = (const float*)d_in[6];
    const float* Wg3 = (const float*)d_in[7];  const float* bg3 = (const float*)d_in[8];
    const float* Wr  = (const float*)d_in[9];  const float* br  = (const float*)d_in[10];
    const float* Wf  = (const float*)d_in[11]; const float* bf  = (const float*)d_in[12];
    const float* Wo  = (const float*)d_in[13]; const float* bo  = (const float*)d_in[14];
    const int* srcI = ei;             // edge_index[0]
    const int* dstI = ei + N_EDGES;   // edge_index[1]
    float* out = (float*)d_out;

    // workspace carve-out (256B aligned)
    char* p = (char*)d_ws;
    auto carve = [&](size_t bytes) -> char* {
        char* r = p; p += (bytes + 255) & ~(size_t)255; return r;
    };
    _Float16* Xh   = (_Float16*)carve((size_t)N_NODES * IN_DIMC * 2);
    _Float16* Wt1  = (_Float16*)carve((size_t)GDIM * IN_DIMC * 2);
    _Float16* Wt2  = (_Float16*)carve((size_t)GDIM * GDIM * 2);
    _Float16* Wt3  = (_Float16*)carve((size_t)GDIM * GDIM * 2);
    _Float16* Wrt  = (_Float16*)carve((size_t)FC0D * CATDIM * 2);
    _Float16* Wft  = (_Float16*)carve((size_t)FC1D * FC0D * 2);
    _Float16* Wot  = (_Float16*)carve((size_t)OUTD * FC1D * 2);
    float*    dinv = (float*)carve((size_t)N_NODES * 4);
    float*    h1   = (float*)carve((size_t)N_NODES * GDIM * 4);
    float*    h2   = (float*)carve((size_t)N_NODES * GDIM * 4);
    float*    h3   = (float*)carve((size_t)N_NODES * GDIM * 4);
    float*    agg  = (float*)carve((size_t)N_NODES * GDIM * 4);
    _Float16* hf16 = (_Float16*)carve((size_t)N_NODES * GDIM * 2);
    float*    g32  = (float*)carve((size_t)NGRAPH * CATDIM * 4);
    _Float16* g16  = (_Float16*)carve((size_t)NGRAPH * CATDIM * 2);
    float*    cnt  = (float*)carve((size_t)NGRAPH * 4);
    _Float16* r1   = (_Float16*)carve((size_t)NGRAPH * FC0D * 2);
    _Float16* r2   = (_Float16*)carve((size_t)NGRAPH * FC1D * 2);

    const int TB = 256;
    auto blocks1d = [](long n, int tb) { return (unsigned)((n + tb - 1) / tb); };

    // ---- precompute: casts, transposed weights, degree norm ----
    cast_f32_to_f16<<<blocks1d((long)N_NODES * IN_DIMC, TB), TB, 0, stream>>>(x, Xh, N_NODES * IN_DIMC);
    transpose_cast<<<blocks1d((long)IN_DIMC * GDIM, TB), TB, 0, stream>>>(Wg1, Wt1, IN_DIMC, GDIM);
    transpose_cast<<<blocks1d((long)GDIM * GDIM, TB), TB, 0, stream>>>(Wg2, Wt2, GDIM, GDIM);
    transpose_cast<<<blocks1d((long)GDIM * GDIM, TB), TB, 0, stream>>>(Wg3, Wt3, GDIM, GDIM);
    transpose_cast<<<blocks1d((long)CATDIM * FC0D, TB), TB, 0, stream>>>(Wr, Wrt, CATDIM, FC0D);
    transpose_cast<<<blocks1d((long)FC0D * FC1D, TB), TB, 0, stream>>>(Wf, Wft, FC0D, FC1D);
    transpose_cast<<<blocks1d((long)FC1D * OUTD, TB), TB, 0, stream>>>(Wo, Wot, FC1D, OUTD);

    deg_init<<<blocks1d(N_NODES, TB), TB, 0, stream>>>(dinv, N_NODES);
    deg_edges<<<blocks1d(N_EDGES, TB), TB, 0, stream>>>(dstI, dinv, N_EDGES);
    deg_rsqrt<<<blocks1d(N_NODES, TB), TB, 0, stream>>>(dinv, N_NODES);

    const unsigned gcnTiles  = (unsigned)((N_NODES / 16) * (GDIM / 16)); // 100000
    const unsigned gcnBlocks = (gcnTiles + 7) / 8;
    const unsigned scatBlks  = blocks1d((long)N_EDGES * 128, TB);
    const unsigned nodeBlks  = blocks1d((long)N_NODES * GDIM, TB);

    // ---- GCN layer 1 ----
    gemm_f16_wmma<<<gcnBlocks, TB, 0, stream>>>(Xh, Wt1, nullptr, h1, nullptr,
                                                N_NODES, IN_DIMC, GDIM, 0);
    hipMemsetAsync(agg, 0, (size_t)N_NODES * GDIM * 4, stream);
    edge_scatter<<<scatBlks, TB, 0, stream>>>(srcI, dstI, dinv, h1, agg);
    gcn_finalize<<<nodeBlks, TB, 0, stream>>>(agg, h1, bg1, dinv, hf16);

    // ---- GCN layer 2 ----
    gemm_f16_wmma<<<gcnBlocks, TB, 0, stream>>>(hf16, Wt2, nullptr, h2, nullptr,
                                                N_NODES, GDIM, GDIM, 0);
    hipMemsetAsync(agg, 0, (size_t)N_NODES * GDIM * 4, stream);
    edge_scatter<<<scatBlks, TB, 0, stream>>>(srcI, dstI, dinv, h2, agg);
    gcn_finalize<<<nodeBlks, TB, 0, stream>>>(agg, h2, bg2, dinv, hf16);

    // ---- GCN layer 3 ----
    gemm_f16_wmma<<<gcnBlocks, TB, 0, stream>>>(hf16, Wt3, nullptr, h3, nullptr,
                                                N_NODES, GDIM, GDIM, 0);
    hipMemsetAsync(agg, 0, (size_t)N_NODES * GDIM * 4, stream);
    edge_scatter<<<scatBlks, TB, 0, stream>>>(srcI, dstI, dinv, h3, agg);
    gcn_finalize<<<nodeBlks, TB, 0, stream>>>(agg, h3, bg3, dinv, nullptr);

    // ---- global mean pool over concat(h1,h2,h3) ----
    hipMemsetAsync(g32, 0, (size_t)NGRAPH * CATDIM * 4, stream);
    hipMemsetAsync(cnt, 0, (size_t)NGRAPH * 4, stream);
    const unsigned poolBlks = blocks1d((long)(N_NODES / 8) * GDIM, TB);
    pool_sum<<<poolBlks, TB, 0, stream>>>(h1, batch, g32, 0);
    pool_sum<<<poolBlks, TB, 0, stream>>>(h2, batch, g32, GDIM);
    pool_sum<<<poolBlks, TB, 0, stream>>>(h3, batch, g32, 2 * GDIM);
    count_nodes<<<blocks1d(N_NODES, TB), TB, 0, stream>>>(batch, cnt, N_NODES);
    pool_div<<<blocks1d((long)NGRAPH * CATDIM, TB), TB, 0, stream>>>(cnt, g32, g16);

    // ---- MLP head (64 rows) ----
    gemm_f16_wmma<<<((NGRAPH / 16) * (FC0D / 16) + 7) / 8, TB, 0, stream>>>(
        g16, Wrt, br, nullptr, r1, NGRAPH, CATDIM, FC0D, 1);
    gemm_f16_wmma<<<((NGRAPH / 16) * (FC1D / 16) + 7) / 8, TB, 0, stream>>>(
        r1, Wft, bf, nullptr, r2, NGRAPH, FC0D, FC1D, 1);
    gemm_f16_wmma<<<((NGRAPH / 16) * (OUTD / 16) + 7) / 8, TB, 0, stream>>>(
        r2, Wot, bo, out, nullptr, NGRAPH, FC1D, OUTD, 0);

    (void)in_sizes; (void)n_in; (void)out_size; (void)ws_size;
}
